// AttentionBlock_17781164605569
// MI455X (gfx1250) — matrix operand order repaired and checked by reading, not compiled, not run on hardware
//
#include <hip/hip_runtime.h>

typedef __attribute__((ext_vector_type(16))) __bf16 v16bf;
typedef __attribute__((ext_vector_type(8)))  float  v8f;

constexpr int kN  = 4096;   // 16*16*16 spatial positions
constexpr int kC  = 256;
constexpr int kCq = 32;
constexpr int kB  = 8;

static __device__ __forceinline__ unsigned short f2bf(float f) {
    union { float f; unsigned int u; } v; v.f = f;
    unsigned int u = v.u;
    if ((u & 0x7fffffffu) > 0x7f800000u) return (unsigned short)((u >> 16) | 0x0040u); // quiet NaN
    return (unsigned short)((u + 0x7fffu + ((u >> 16) & 1u)) >> 16);                   // RNE
}

union Frag16 { uint4 u4[2]; unsigned int u[8]; v16bf v; };

// ---------------------------------------------------------------------------
// Kernel 0: pack wq/wk/wv into one bf16 weight matrix wall[320][256]
// (rows 0-31 = wq, 32-63 = wk, 64-319 = wv) + fused f32 bias ball[320].
// Row-major [o][c] is exactly the WMMA A-fragment layout (two contiguous
// 16B K-chunks per lane).
// ---------------------------------------------------------------------------
__global__ __launch_bounds__(256) void wcvt_kernel(
    const float* __restrict__ wq, const float* __restrict__ bq,
    const float* __restrict__ wk, const float* __restrict__ bk,
    const float* __restrict__ wv, const float* __restrict__ bv,
    unsigned short* __restrict__ wall, float* __restrict__ ball) {
  const int o = blockIdx.x, tid = threadIdx.x;
  const float* src; float bias;
  if (o < 32)      { src = wq + o * kC;        bias = bq[o];      }
  else if (o < 64) { src = wk + (o - 32) * kC; bias = bk[o - 32]; }
  else             { src = wv + (o - 64) * kC; bias = bv[o - 64]; }
  wall[o * kC + tid] = f2bf(src[tid]);
  if (tid == 0) ball[o] = bias;
}

// ---------------------------------------------------------------------------
// Kernel 1: QKV projection as a WMMA GEMM: [320x256] x [256 x 32n] per strip.
// x strip is staged transposed in LDS as bf16 (padded rows -> conflict-free
// ds_load_b128 B-fragments). Each of 8 waves computes 5 of the 40 16x16
// output tiles with 8 K-steps of v_wmma_f32_16x16x32_bf16. Outputs written
// in the layouts the attention kernel's fragments want:
//   q  [B][N][32], kT [B][N][32], v [B][C][N]   (all bf16)
// ---------------------------------------------------------------------------
__global__ __launch_bounds__(256) void qkv_wmma_kernel(
    const float* __restrict__ x,
    const unsigned short* __restrict__ wall, const float* __restrict__ ball,
    unsigned short* __restrict__ qb, unsigned short* __restrict__ ktb,
    unsigned short* __restrict__ vb) {
  constexpr int XP = 264;                 // padded xT row stride (bf16 elems)
  __shared__ unsigned short xT[32 * XP];  // ~16.5 KB
  const int tid  = threadIdx.x;
  const int b    = blockIdx.y;
  const int n0   = blockIdx.x * 32;

  // stage + transpose + convert x[b][c][n0..n0+31] -> xT[n][c]
  for (int idx = tid; idx < kC * 32; idx += 256) {
    int c = idx >> 5, n = idx & 31;
    xT[n * XP + c] = f2bf(x[((size_t)(b * kC + c)) * kN + n0 + n]);
  }
  __syncthreads();

  const int w    = tid >> 5;
  const int lane = tid & 31;
  const int half = lane >> 4;
  const int lm   = lane & 15;

  v8f zero8;
  #pragma unroll
  for (int r = 0; r < 8; ++r) zero8[r] = 0.0f;

  #pragma unroll
  for (int s = 0; s < 5; ++s) {
    const int tt = w * 5 + s;             // 0..39
    const int ot = tt >> 1;               // o-tile 0..19
    const int nt = tt & 1;                // n-subtile 0..1
    v8f acc = zero8;
    #pragma unroll
    for (int ks = 0; ks < 8; ++ks) {
      Frag16 af;                          // A: weights, row o, K chunk pair
      const unsigned short* wrow = wall + (size_t)(ot * 16 + lm) * kC + ks * 32 + half * 8;
      af.u4[0] = *(const uint4*)(wrow);
      af.u4[1] = *(const uint4*)(wrow + 16);
      Frag16 bf_;                         // B: xT from LDS, col n, K chunk pair
      const unsigned short* xrow = xT + (nt * 16 + lm) * XP + ks * 32 + half * 8;
      bf_.u4[0] = *(const uint4*)(xrow);
      bf_.u4[1] = *(const uint4*)(xrow + 16);
      acc = __builtin_amdgcn_wmma_f32_16x16x32_bf16(false, af.v, false, bf_.v,
                                                    (short)0, acc, false, false);
    }
    // epilogue: bias + cast + store (D layout: M = r + 8*half, N = lm)
    const int n = n0 + nt * 16 + lm;
    float val[8];
    #pragma unroll
    for (int r = 0; r < 8; ++r) val[r] = acc[r] + ball[ot * 16 + 8 * half + r];
    if (ot < 2) {                         // q or kT: 8 consecutive o -> b128
      uint4 p;
      p.x = (unsigned int)f2bf(val[0]) | ((unsigned int)f2bf(val[1]) << 16);
      p.y = (unsigned int)f2bf(val[2]) | ((unsigned int)f2bf(val[3]) << 16);
      p.z = (unsigned int)f2bf(val[4]) | ((unsigned int)f2bf(val[5]) << 16);
      p.w = (unsigned int)f2bf(val[6]) | ((unsigned int)f2bf(val[7]) << 16);
      unsigned short* dst = (ot == 0 ? qb : ktb);
      *(uint4*)(dst + ((size_t)(b * kN + n)) * kCq + 8 * half) = p;
    } else {                              // v[c][n]: per-lane scattered u16
      #pragma unroll
      for (int r = 0; r < 8; ++r) {
        const int c = ot * 16 + 8 * half + r - 64;
        vb[((size_t)(b * kC + c)) * kN + n] = f2bf(val[r]);
      }
    }
  }
}

// ---------------------------------------------------------------------------
// Kernel 2: flash attention for one (batch, 16-row i-block).
// 8 waves; each wave owns a strided j-subset and the full C=256 output width
// (16 f32 accumulator tiles = 128 VGPRs). Two passes: global row-max, then
// exp + P·V with WMMA. Sᵀ is computed (M=j, N=i) so softmax stats are
// per-lane. P is rearranged C-layout -> B-layout fully in-register via
// shfl_xor(16) + bf16 packing (the lane<->N mapping is shared by both).
// ---------------------------------------------------------------------------
__global__ __launch_bounds__(256) void attn_kernel(
    const unsigned short* __restrict__ qb, const unsigned short* __restrict__ ktb,
    const unsigned short* __restrict__ vb, const float* __restrict__ x,
    float* __restrict__ rout) {
  constexpr int OFP = 257;               // pad to dodge LDS bank conflicts
  __shared__ float Of[16 * OFP];
  __shared__ float red[128];
  __shared__ float linv[16];
  const int tid  = threadIdx.x;
  const int w    = tid >> 5;
  const int lane = tid & 31;
  const int half = lane >> 4;            // lane-half selects K/M range
  const int lm   = lane & 15;            // N index (= softmax row i)
  const int b    = blockIdx.y;
  const int i0   = blockIdx.x * 16;

  for (int e = tid; e < 16 * OFP; e += 256) Of[e] = 0.0f;

  v8f zero8;
  #pragma unroll
  for (int r = 0; r < 8; ++r) zero8[r] = 0.0f;

  // persistent q B-fragment: lane lm = column i, K(=o) = half*8 .. +7, 16+half*8 .. +7
  Frag16 qf;
  {
    const unsigned short* qrow = qb + ((size_t)(b * kN + i0 + lm)) * kCq + half * 8;
    qf.u4[0] = *(const uint4*)(qrow);
    qf.u4[1] = *(const uint4*)(qrow + 16);
  }

  // ---- pass 1: global max of logits per row i (per-lane, N = lm) ----
  float m = -3.0e38f;
  for (int t = 0; t < 32; ++t) {
    const int j0 = (w + 8 * t) * 16;
    Frag16 kf;
    const unsigned short* krow = ktb + ((size_t)(b * kN + j0 + lm)) * kCq + half * 8;
    kf.u4[0] = *(const uint4*)(krow);
    kf.u4[1] = *(const uint4*)(krow + 16);
    v8f s = __builtin_amdgcn_wmma_f32_16x16x32_bf16(false, kf.v, false, qf.v,
                                                    (short)0, zero8, false, false);
    #pragma unroll
    for (int r = 0; r < 8; ++r) m = fmaxf(m, s[r]);
  }
  m = fmaxf(m, __shfl_xor(m, 16, 32));
  if (lane < 16) red[w * 16 + lane] = m;
  __syncthreads();
  float gmax = red[lm];
  #pragma unroll
  for (int w2 = 1; w2 < 8; ++w2) gmax = fmaxf(gmax, red[w2 * 16 + lm]);
  __syncthreads();                       // red will be reused for l

  // ---- pass 2: P = exp(scale*(S - max)), O += P·V ----
  v8f acc[16];
  #pragma unroll
  for (int ct = 0; ct < 16; ++ct) acc[ct] = zero8;
  float lp = 0.0f;
  const float sc = 0.0625f * 1.4426950408889634f;   // (1/sqrt(C)) * log2(e)

  for (int t = 0; t < 16; ++t) {
    const int j0 = (w + 8 * t) * 32;
    Frag16 ka, kb2;
    const unsigned short* krowA = ktb + ((size_t)(b * kN + j0 + lm)) * kCq + half * 8;
    const unsigned short* krowB = krowA + 16 * kCq;
    ka.u4[0]  = *(const uint4*)(krowA);  ka.u4[1]  = *(const uint4*)(krowA + 16);
    kb2.u4[0] = *(const uint4*)(krowB);  kb2.u4[1] = *(const uint4*)(krowB + 16);
    v8f sa = __builtin_amdgcn_wmma_f32_16x16x32_bf16(false, ka.v,  false, qf.v,
                                                     (short)0, zero8, false, false);
    v8f sb = __builtin_amdgcn_wmma_f32_16x16x32_bf16(false, kb2.v, false, qf.v,
                                                     (short)0, zero8, false, false);
    float Ea[8], Eb[8], oa[8], ob[8];
    #pragma unroll
    for (int r = 0; r < 8; ++r) {
      Ea[r] = exp2f((sa[r] - gmax) * sc);
      Eb[r] = exp2f((sb[r] - gmax) * sc);
      lp += Ea[r] + Eb[r];
    }
    #pragma unroll
    for (int r = 0; r < 8; ++r) {
      oa[r] = __shfl_xor(Ea[r], 16, 32);
      ob[r] = __shfl_xor(Eb[r], 16, 32);
    }
    // C-layout (M = r + 8*half, N = lm) -> B-layout (K pairs per VGPR, N = lm)
    // lanes<16 need tile-a K=0..15; lanes>=16 need tile-b K=16..31.
    float lo[16];
    #pragma unroll
    for (int r = 0; r < 8; ++r) {
      lo[r]     = half ? ob[r] : Ea[r];
      lo[8 + r] = half ? Eb[r] : oa[r];
    }
    Frag16 pb;
    #pragma unroll
    for (int r = 0; r < 8; ++r)
      pb.u[r] = (unsigned int)f2bf(lo[2 * r]) | ((unsigned int)f2bf(lo[2 * r + 1]) << 16);

    #pragma unroll
    for (int ct = 0; ct < 16; ++ct) {
      Frag16 vf;
      const unsigned short* vrow =
          vb + ((size_t)(b * kC + ct * 16 + lm)) * kN + j0 + half * 16;
      vf.u4[0] = *(const uint4*)(vrow);
      vf.u4[1] = *(const uint4*)(vrow + 8);
      acc[ct] = __builtin_amdgcn_wmma_f32_16x16x32_bf16(false, vf.v, false, pb.v,
                                                        (short)0, acc[ct], false, false);
    }
  }

  // ---- combine across waves ----
  lp += __shfl_xor(lp, 16, 32);
  if (lane < 16) red[w * 16 + lane] = lp;
  __syncthreads();
  if (tid < 16) {
    float s = 0.0f;
    for (int w2 = 0; w2 < 8; ++w2) s += red[w2 * 16 + tid];
    linv[tid] = 1.0f / s;
  }
  #pragma unroll
  for (int ct = 0; ct < 16; ++ct) {
    #pragma unroll
    for (int r = 0; r < 8; ++r)
      atomicAdd(&Of[lm * OFP + ct * 16 + r + 8 * half], acc[ct][r]);
  }
  __syncthreads();
  for (int e = tid; e < 16 * kC; e += 256) {
    int i = e & 15, c = e >> 4;
    size_t idx = ((size_t)(b * kC + c)) * kN + i0 + i;
    rout[idx] = Of[i * OFP + c] * linv[i] + x[idx];      // residual
  }
}

// ---------------------------------------------------------------------------
// Kernel 3: per-channel batch statistics (training-mode, biased variance)
// ---------------------------------------------------------------------------
__global__ __launch_bounds__(256) void bnstats_kernel(const float* __restrict__ r,
                                                      float* __restrict__ stats) {
  __shared__ float s1[256], s2[256];
  const int c = blockIdx.x, tid = threadIdx.x;
  float a = 0.0f, q = 0.0f;
  for (int idx = tid; idx < kB * kN; idx += 256) {
    int b = idx >> 12, n = idx & (kN - 1);
    float v = r[((size_t)(b * kC + c)) * kN + n];
    a += v; q += v * v;
  }
  s1[tid] = a; s2[tid] = q;
  __syncthreads();
  for (int s = 128; s > 0; s >>= 1) {
    if (tid < s) { s1[tid] += s1[tid + s]; s2[tid] += s2[tid + s]; }
    __syncthreads();
  }
  if (tid == 0) {
    const float inv = 1.0f / (float)(kB * kN);
    float mean = s1[0] * inv;
    float var  = s2[0] * inv - mean * mean;
    stats[c]      = mean;
    stats[kC + c] = rsqrtf(var + 1e-5f);
  }
}

// ---------------------------------------------------------------------------
// Kernel 4: in-place normalize + affine + ReLU
// ---------------------------------------------------------------------------
__global__ __launch_bounds__(256) void bnapply_kernel(float* __restrict__ r,
    const float* __restrict__ stats, const float* __restrict__ gamma,
    const float* __restrict__ beta) {
  size_t idx = (size_t)blockIdx.x * 256 + threadIdx.x;
  int c = (int)((idx >> 12) & (kC - 1));
  float v = r[idx];
  float y = (v - stats[c]) * stats[kC + c] * gamma[c] + beta[c];
  r[idx] = fmaxf(y, 0.0f);
}

extern "C" void kernel_launch(void* const* d_in, const int* in_sizes, int n_in,
                              void* d_out, int out_size, void* d_ws, size_t ws_size,
                              hipStream_t stream) {
  (void)in_sizes; (void)n_in; (void)out_size; (void)ws_size;
  const float* x     = (const float*)d_in[0];
  const float* wq    = (const float*)d_in[1];
  const float* bq    = (const float*)d_in[2];
  const float* wk    = (const float*)d_in[3];
  const float* bk    = (const float*)d_in[4];
  const float* wv    = (const float*)d_in[5];
  const float* bv    = (const float*)d_in[6];
  const float* gamma = (const float*)d_in[7];
  const float* beta  = (const float*)d_in[8];
  float* out = (float*)d_out;

  // workspace layout (bf16 staging ~20 MB + stats + packed weights)
  unsigned short* qb   = (unsigned short*)d_ws;
  unsigned short* ktb  = qb  + (size_t)kB * kN * kCq;       // +2 MB
  unsigned short* vb   = ktb + (size_t)kB * kN * kCq;       // +2 MB
  float*          stats = (float*)(vb + (size_t)kB * kC * kN); // +16 MB
  unsigned short* wall = (unsigned short*)(stats + 2 * kC); // 320x256 bf16
  float*          ball = (float*)(wall + 320 * kC);         // 320 f32

  wcvt_kernel<<<320, 256, 0, stream>>>(wq, bq, wk, bk, wv, bv, wall, ball);
  qkv_wmma_kernel<<<dim3(kN / 32, kB), 256, 0, stream>>>(x, wall, ball, qb, ktb, vb);
  attn_kernel<<<dim3(kN / 16, kB), 256, 0, stream>>>(qb, ktb, vb, x, out);
  bnstats_kernel<<<kC, 256, 0, stream>>>(out, stats);
  bnapply_kernel<<<(kB * kC * kN) / 256, 256, 0, stream>>>(out, stats, gamma, beta);
}
